// InvariantPointAttention_13932873908955
// MI455X (gfx1250) — compile-verified, matrix operand order checked by reading, not run on hardware
//
#include <hip/hip_runtime.h>
#include <hip/hip_bf16.h>
#include <math.h>

// ---------------- CDNA5 WMMA plumbing (wave32, 16x16x32 bf16) ----------------
typedef __attribute__((ext_vector_type(16))) __bf16 v16bf;
typedef __attribute__((ext_vector_type(8)))  __bf16 v8bf;
typedef __attribute__((ext_vector_type(8)))  float  v8f;
typedef __attribute__((ext_vector_type(4)))  float  v4f;

#define LQ 512
#define NH 8
#define DHD 48

// 16-bit operand per-lane K layout: kk = (e&7) + (e>>3)*16 + g*8  (g = lane>>4)
// => two contiguous 8-element runs at k0+8g and k0+16+8g.
__device__ __forceinline__ int kmap16(int e, int g) {
  return (e & 7) + ((e >> 3) << 4) + (g << 3);
}

__device__ __forceinline__ v8f wmma_bf16(v16bf a, v16bf b, v8f c) {
  return __builtin_amdgcn_wmma_f32_16x16x32_bf16(false, a, false, b, (short)0, c,
                                                 false, false);
}

// Load 16 fp32 (two contiguous float4-pairs) and convert to bf16 operand.
__device__ __forceinline__ v16bf load_a_f32(const float* __restrict__ p, int g) {
  const v4f* q0 = (const v4f*)(p + g * 8);
  const v4f* q1 = (const v4f*)(p + 16 + g * 8);
  v4f x0 = q0[0], x1 = q0[1], y0 = q1[0], y1 = q1[1];
  v16bf out;
#pragma unroll
  for (int i = 0; i < 4; ++i) {
    out[i]      = (__bf16)x0[i];
    out[4 + i]  = (__bf16)x1[i];
    out[8 + i]  = (__bf16)y0[i];
    out[12 + i] = (__bf16)y1[i];
  }
  return out;
}

// Load 16 bf16 (two contiguous 16B runs) as an operand.
__device__ __forceinline__ v16bf load_row_bf16(const __bf16* __restrict__ p, int g) {
  v8bf lo = *(const v8bf*)(p + g * 8);
  v8bf hi = *(const v8bf*)(p + 16 + g * 8);
  return __builtin_shufflevector(lo, hi, 0, 1, 2, 3, 4, 5, 6, 7,
                                 8, 9, 10, 11, 12, 13, 14, 15);
}

// ---------------- generic fp32 GEMM (convert->bf16 WMMA), C = A@B + bias -----
// A [M,K] row-major fp32, B [K,N] row-major fp32, M,N multiples of 16, K of 32.
__global__ void gemm_f32_wmma(const float* __restrict__ A, const float* __restrict__ B,
                              const float* __restrict__ bias, float* __restrict__ C,
                              int M, int N, int K) {
  int lane = threadIdx.x & 31;
  int wave = threadIdx.x >> 5;
  int ntiles = N >> 4;
  int tile = blockIdx.x * (blockDim.x >> 5) + wave;
  if (tile >= (M >> 4) * ntiles) return;
  int mt = tile / ntiles, nt = tile % ntiles;
  int g = lane >> 4;
  int row = mt * 16 + (lane & 15);
  int col = nt * 16 + (lane & 15);
  const float* ap = A + (size_t)row * K;
  v8f acc = {};
  for (int k0 = 0; k0 < K; k0 += 32) {
    v16bf a = load_a_f32(ap + k0, g);
    v16bf b;
#pragma unroll
    for (int e = 0; e < 16; ++e)
      b[e] = (__bf16)B[(size_t)(k0 + kmap16(e, g)) * N + col];  // coalesced across lanes
    acc = wmma_bf16(a, b, acc);
  }
  float bv = bias ? bias[col] : 0.0f;
#pragma unroll
  for (int r = 0; r < 8; ++r) {
    int m = mt * 16 + r + 8 * g;
    C[(size_t)m * N + col] = acc[r] + bv;
  }
}

// ---------------- fused z projection: bias(N=8 pad 16) + pair_z(N=32) --------
// z [LL,128] read exactly once (dominant 134MB HBM traffic). Persistent waves:
// all 12 weight operands (3 N-tiles x 4 k-steps) are hoisted into registers and
// reused across the grid-stride tile loop; outputs stored directly as bf16.
#define ZWAVES 4096
__global__ void zproj_kernel(const float* __restrict__ z, const float* __restrict__ Wb,
                             const float* __restrict__ bb, const float* __restrict__ Wdown,
                             const float* __restrict__ bdown,
                             __bf16* __restrict__ bias16, __bf16* __restrict__ pairz) {
  int lane = threadIdx.x & 31;
  int wave = threadIdx.x >> 5;
  int mt0 = blockIdx.x * (blockDim.x >> 5) + wave;
  int g = lane >> 4;
  int col = lane & 15;
  // hoisted weight operands
  v16bf B0[4], B1[4], B2[4];
#pragma unroll
  for (int ks = 0; ks < 4; ++ks) {
#pragma unroll
    for (int e = 0; e < 16; ++e) {
      int kk = ks * 32 + kmap16(e, g);
      B0[ks][e] = (col < 8) ? (__bf16)Wb[(size_t)kk * 8 + col] : (__bf16)0.0f;
      B1[ks][e] = (__bf16)Wdown[(size_t)kk * 32 + col];
      B2[ks][e] = (__bf16)Wdown[(size_t)kk * 32 + 16 + col];
    }
  }
  float bbv = (col < 8) ? bb[col] : 0.0f;
  float bd0 = bdown[col], bd1 = bdown[16 + col];
  for (int mt = mt0; mt < (LQ * LQ) / 16; mt += ZWAVES) {
    int row = mt * 16 + (lane & 15);
    const float* ap = z + (size_t)row * 128;
    __builtin_prefetch(ap + (size_t)ZWAVES * 16 * 128, 0, 1);  // next stride tile
    v8f accB = {}, accP0 = {}, accP1 = {};
#pragma unroll
    for (int ks = 0; ks < 4; ++ks) {
      v16bf a = load_a_f32(ap + ks * 32, g);
      accB  = wmma_bf16(a, B0[ks], accB);
      accP0 = wmma_bf16(a, B1[ks], accP0);
      accP1 = wmma_bf16(a, B2[ks], accP1);
    }
#pragma unroll
    for (int r = 0; r < 8; ++r) {
      int m = mt * 16 + r + 8 * g;
      bias16[(size_t)m * 16 + col] = (__bf16)(accB[r] + bbv);
      pairz[(size_t)m * 32 + col] = (__bf16)(accP0[r] + bd0);
      pairz[(size_t)m * 32 + 16 + col] = (__bf16)(accP1[r] + bd1);
    }
  }
}

// ---------------- prep: RoPE, rigid transforms, augmented Q'/K' (dim 64) -----
// logit = (q/12)·k + hw*Sum qp·kp + (-.5hw|qp|^2)*1 + 1*(-.5hw|kp|^2)  [+ bias]
// V and V_pts are written TRANSPOSED [h][d][m] so attention-apply B-loads are
// per-lane contiguous vector loads.
__global__ void prep_kernel(const float* __restrict__ q_lin, const float* __restrict__ kv_lin,
                            const float* __restrict__ qp_lin, const float* __restrict__ kvp_lin,
                            const float* __restrict__ affines, const int* __restrict__ node_pos,
                            const float* __restrict__ head_weights,
                            __bf16* __restrict__ Qaug, __bf16* __restrict__ Kaug,
                            __bf16* __restrict__ VT, __bf16* __restrict__ VpT) {
  int idx = blockIdx.x * blockDim.x + threadIdx.x;
  if (idx >= LQ * NH) return;
  int l = idx >> 3, h = idx & 7;
  const float c1 = 0.0833333333f;                            // 1/sqrt(3*48)
  float hw = log1pf(__expf(head_weights[h])) * 0.13608276f;  // softplus*sqrt(1/54)
  float pos = (float)node_pos[l];
  __bf16* Q  = Qaug + ((size_t)h * LQ + l) * 64;
  __bf16* Kg = Kaug + ((size_t)h * LQ + l) * 64;
  const float* q  = q_lin  + (size_t)l * (NH * DHD) + h * DHD;
  const float* kv = kv_lin + (size_t)l * (2 * NH * DHD) + h * (2 * DHD);
  for (int i = 0; i < 24; ++i) {
    float ang = pos * __expf(-(float)i * 0.38376418f);  // ln(1e4)/24
    float cs = __cosf(ang), sn = __sinf(ang);
    float q1 = q[i], q2 = q[24 + i];
    Q[i]      = (__bf16)(c1 * (q1 * cs - q2 * sn));
    Q[24 + i] = (__bf16)(c1 * (q1 * sn + q2 * cs));
    float k1 = kv[i], k2 = kv[24 + i];
    Kg[i]      = (__bf16)(k1 * cs - k2 * sn);
    Kg[24 + i] = (__bf16)(k1 * sn + k2 * cs);
  }
  for (int d = 0; d < DHD; ++d)
    VT[((size_t)h * DHD + d) * LQ + l] = (__bf16)kv[DHD + d];
  const float* af = affines + (size_t)l * 12;
  float R00 = af[0], R01 = af[1], R02 = af[2],  T0 = af[3];
  float R10 = af[4], R11 = af[5], R12 = af[6],  T1 = af[7];
  float R20 = af[8], R21 = af[9], R22 = af[10], T2 = af[11];
  float qn = 0.f;
  for (int p = 0; p < 4; ++p) {
    int n = h * 4 + p;   // _split3: [l, c*32+n]
    float x = qp_lin[(size_t)l * 96 + n];
    float y = qp_lin[(size_t)l * 96 + 32 + n];
    float zc = qp_lin[(size_t)l * 96 + 64 + n];
    float gx = R00 * x + R01 * y + R02 * zc + T0;
    float gy = R10 * x + R11 * y + R12 * zc + T1;
    float gz = R20 * x + R21 * y + R22 * zc + T2;
    Q[48 + p * 3 + 0] = (__bf16)(hw * gx);
    Q[48 + p * 3 + 1] = (__bf16)(hw * gy);
    Q[48 + p * 3 + 2] = (__bf16)(hw * gz);
    qn += gx * gx + gy * gy + gz * gz;
  }
  Q[60] = (__bf16)(-0.5f * hw * qn);
  Q[61] = (__bf16)1.0f;
  Q[62] = (__bf16)0.0f;
  Q[63] = (__bf16)0.0f;
  float kn = 0.f;
  for (int p = 0; p < 12; ++p) {
    int n = h * 12 + p;  // _split3: [l, c*96+n]
    float x = kvp_lin[(size_t)l * 288 + n];
    float y = kvp_lin[(size_t)l * 288 + 96 + n];
    float zc = kvp_lin[(size_t)l * 288 + 192 + n];
    float gx = R00 * x + R01 * y + R02 * zc + T0;
    float gy = R10 * x + R11 * y + R12 * zc + T1;
    float gz = R20 * x + R21 * y + R22 * zc + T2;
    if (p < 4) {
      Kg[48 + p * 3 + 0] = (__bf16)gx;
      Kg[48 + p * 3 + 1] = (__bf16)gy;
      Kg[48 + p * 3 + 2] = (__bf16)gz;
      kn += gx * gx + gy * gy + gz * gz;
    } else {
      VpT[((size_t)h * 24 + (p - 4) * 3 + 0) * LQ + l] = (__bf16)gx;
      VpT[((size_t)h * 24 + (p - 4) * 3 + 1) * LQ + l] = (__bf16)gy;
      VpT[((size_t)h * 24 + (p - 4) * 3 + 2) * LQ + l] = (__bf16)gz;
    }
  }
  Kg[60] = (__bf16)1.0f;
  Kg[61] = (__bf16)(-0.5f * hw * kn);
  Kg[62] = (__bf16)0.0f;
  Kg[63] = (__bf16)0.0f;
}

// ---------------- attention logits: Q'[h] @ K'[h]^T (K=64) + c2*bias ---------
__global__ void logits_kernel(const __bf16* __restrict__ Qaug, const __bf16* __restrict__ Kaug,
                              const __bf16* __restrict__ bias16, float* __restrict__ logits) {
  int lane = threadIdx.x & 31;
  int wave = threadIdx.x >> 5;
  int tile = blockIdx.x * (blockDim.x >> 5) + wave;  // 8*32*32 tiles
  if (tile >= NH * 32 * 32) return;
  int h = tile >> 10, lt = (tile >> 5) & 31, mt = tile & 31;
  int g = lane >> 4;
  const __bf16* Qp = Qaug + ((size_t)h * LQ + lt * 16 + (lane & 15)) * 64;
  const __bf16* Kp = Kaug + ((size_t)h * LQ + mt * 16 + (lane & 15)) * 64;  // B = K'^T
  v8f acc = {};
#pragma unroll
  for (int k0 = 0; k0 < 64; k0 += 32) {
    v16bf a = load_row_bf16(Qp + k0, g);
    v16bf b = load_row_bf16(Kp + k0, g);
    acc = wmma_bf16(a, b, acc);
  }
  const float c2 = 0.57735027f;  // sqrt(1/3)
  int m = mt * 16 + (lane & 15);
#pragma unroll
  for (int r = 0; r < 8; ++r) {
    int l = lt * 16 + r + 8 * g;
    logits[((size_t)h * LQ + l) * LQ + m] =
        acc[r] + c2 * (float)bias16[((size_t)l * LQ + m) * 16 + h];
  }
}

// ---------------- softmax over m, one wave per (h,l) row ---------------------
__global__ void softmax_kernel(const float* __restrict__ logits, const float* __restrict__ mask,
                               __bf16* __restrict__ a_bf) {
  int lane = threadIdx.x & 31;
  int wave = threadIdx.x >> 5;
  int row = blockIdx.x * (blockDim.x >> 5) + wave;  // 8*512 rows
  if (row >= NH * LQ) return;
  int l = row & (LQ - 1);
  const float* x = logits + (size_t)row * LQ;
  float ml = mask[l];
  float vals[16];
  float vmax = -3.4e38f;
#pragma unroll
  for (int i = 0; i < 16; ++i) {
    int m = lane + i * 32;
    float v = x[m] + 1.0e6f * (ml * mask[m] - 1.0f);
    vals[i] = v;
    vmax = fmaxf(vmax, v);
  }
  for (int off = 16; off; off >>= 1) vmax = fmaxf(vmax, __shfl_xor(vmax, off, 32));
  float s = 0.f;
#pragma unroll
  for (int i = 0; i < 16; ++i) { vals[i] = __expf(vals[i] - vmax); s += vals[i]; }
  for (int off = 16; off; off >>= 1) s += __shfl_xor(s, off, 32);
  float inv = 1.0f / s;
#pragma unroll
  for (int i = 0; i < 16; ++i)
    a_bf[(size_t)row * LQ + lane + i * 32] = (__bf16)(vals[i] * inv);
}

// ------- attention apply: C[l, h*chead+n] = sum_m A[h,l,m] * VT[h,n,m] -------
__global__ void av_kernel(const __bf16* __restrict__ Aall, const __bf16* __restrict__ BTall,
                          float* __restrict__ C, int Nc, int ldc, int chead) {
  int lane = threadIdx.x & 31;
  int wave = threadIdx.x >> 5;
  int h = blockIdx.y;
  int ntiles = (Nc + 15) >> 4;
  int tile = blockIdx.x * (blockDim.x >> 5) + wave;
  if (tile >= 32 * ntiles) return;
  int lt = tile / ntiles, nt = tile % ntiles;
  int g = lane >> 4;
  int col = nt * 16 + (lane & 15);
  bool cok = col < Nc;
  int colc = cok ? col : (Nc - 1);  // clamp: invalid cols computed but not stored
  const __bf16* ap = Aall + ((size_t)h * LQ + lt * 16 + (lane & 15)) * LQ;
  const __bf16* bp = BTall + ((size_t)h * Nc + colc) * LQ;
  v8f acc = {};
  for (int k0 = 0; k0 < LQ; k0 += 32) {
    v16bf a = load_row_bf16(ap + k0, g);
    v16bf b = load_row_bf16(bp + k0, g);
    acc = wmma_bf16(a, b, acc);
  }
  if (cok) {
#pragma unroll
    for (int r = 0; r < 8; ++r) {
      int l = lt * 16 + r + 8 * g;
      C[(size_t)l * ldc + h * chead + col] = acc[r];
    }
  }
}

// ---------------- o_pair: per l, A rows = heads (8, pad 16), K=512, N=32 -----
__global__ void opair_kernel(const __bf16* __restrict__ a_bf, const __bf16* __restrict__ pairz,
                             float* __restrict__ feats) {
  int lane = threadIdx.x & 31;
  int wave = threadIdx.x >> 5;
  int l = blockIdx.x * (blockDim.x >> 5) + wave;
  if (l >= LQ) return;
  int g = lane >> 4;
  int hrow = lane & 15;
  int col = lane & 15;
  const __bf16* Z = pairz + (size_t)l * LQ * 32;
  const __bf16* ap = a_bf + ((size_t)(hrow & 7) * LQ + l) * LQ;  // clamped in-bounds
  bool hok = hrow < 8;
  v8f acc0 = {}, acc1 = {};
  for (int k0 = 0; k0 < LQ; k0 += 32) {
    v16bf a = load_row_bf16(ap + k0, g);
    if (!hok) a = (v16bf)((__bf16)0.0f);   // zero padded head rows
    v16bf b0, b1;
#pragma unroll
    for (int e = 0; e < 16; ++e) {
      int kk = k0 + kmap16(e, g);
      b0[e] = Z[(size_t)kk * 32 + col];        // coalesced across lanes
      b1[e] = Z[(size_t)kk * 32 + 16 + col];
    }
    acc0 = wmma_bf16(a, b0, acc0);
    acc1 = wmma_bf16(a, b1, acc1);
  }
#pragma unroll
  for (int r = 0; r < 8; ++r) {
    int h = r + 8 * g;
    if (h < 8) {
      feats[(size_t)l * 896 + 640 + h * 32 + col] = acc0[r];
      feats[(size_t)l * 896 + 640 + h * 32 + 16 + col] = acc1[r];
    }
  }
}

// ---------------- o_pt: inverse frame, norms, write into feats ---------------
__global__ void optassemble_kernel(const float* __restrict__ optg, const float* __restrict__ affines,
                                   float* __restrict__ feats) {
  int idx = blockIdx.x * blockDim.x + threadIdx.x;
  if (idx >= LQ * 64) return;
  int l = idx >> 6, j = idx & 63;        // j = h*NV + v
  int h = j >> 3, v = j & 7;
  const float* af = affines + (size_t)l * 12;
  const float* gsrc = optg + (size_t)l * 192 + h * 24 + v * 3;
  float gx = gsrc[0] - af[3], gy = gsrc[1] - af[7], gz = gsrc[2] - af[11];
  // out_i = sum_j R[j][i] * (o_j - t_j)
  float lx = af[0] * gx + af[4] * gy + af[8]  * gz;
  float ly = af[1] * gx + af[5] * gy + af[9]  * gz;
  float lz = af[2] * gx + af[6] * gy + af[10] * gz;
  float nrm = sqrtf(lx * lx + ly * ly + lz * lz + 1.0e-6f);
  float* F = feats + (size_t)l * 896;
  F[384 + j]       = lx;
  F[384 + 64 + j]  = ly;
  F[384 + 128 + j] = lz;
  F[384 + 192 + j] = nrm;
}

// ---------------- host side -------------------------------------------------
extern "C" void kernel_launch(void* const* d_in, const int* in_sizes, int n_in,
                              void* d_out, int out_size, void* d_ws, size_t ws_size,
                              hipStream_t stream) {
  const float* s         = (const float*)d_in[0];
  const float* z         = (const float*)d_in[1];
  const float* affines   = (const float*)d_in[2];
  const int*   node_pos  = (const int*)d_in[3];
  const float* mask      = (const float*)d_in[4];
  const float* Wq        = (const float*)d_in[5];
  const float* bq        = (const float*)d_in[6];
  const float* Wkv       = (const float*)d_in[7];
  const float* bkv       = (const float*)d_in[8];
  const float* Wqp       = (const float*)d_in[9];
  const float* bqp       = (const float*)d_in[10];
  const float* Wkvp      = (const float*)d_in[11];
  const float* bkvp      = (const float*)d_in[12];
  const float* Wb        = (const float*)d_in[13];
  const float* bb        = (const float*)d_in[14];
  const float* Wdown     = (const float*)d_in[15];
  const float* bdown     = (const float*)d_in[16];
  const float* head_w    = (const float*)d_in[17];
  const float* Wout      = (const float*)d_in[18];
  const float* bout      = (const float*)d_in[19];
  float* out = (float*)d_out;

  char* w = (char*)d_ws;
  auto alloc = [&](size_t bytes) -> void* {
    void* p = (void*)w;
    w += (bytes + 255) & ~(size_t)255;
    return p;
  };
  float*  q_lin   = (float*)alloc((size_t)LQ * 384 * 4);
  float*  kv_lin  = (float*)alloc((size_t)LQ * 768 * 4);
  float*  qp_lin  = (float*)alloc((size_t)LQ * 96 * 4);
  float*  kvp_lin = (float*)alloc((size_t)LQ * 288 * 4);
  __bf16* Qaug    = (__bf16*)alloc((size_t)NH * LQ * 64 * 2);
  __bf16* Kaug    = (__bf16*)alloc((size_t)NH * LQ * 64 * 2);
  __bf16* VT      = (__bf16*)alloc((size_t)NH * 48 * LQ * 2);
  __bf16* VpT     = (__bf16*)alloc((size_t)NH * 24 * LQ * 2);
  __bf16* bias16  = (__bf16*)alloc((size_t)LQ * LQ * 16 * 2);
  __bf16* pairz   = (__bf16*)alloc((size_t)LQ * LQ * 32 * 2);
  float*  logits  = (float*)alloc((size_t)NH * LQ * LQ * 4);
  __bf16* a_bf    = (__bf16*)alloc((size_t)NH * LQ * LQ * 2);
  float*  optg    = (float*)alloc((size_t)LQ * 192 * 4);
  float*  feats   = (float*)alloc((size_t)LQ * 896 * 4);

  const int BLK = 128;  // 4 waves per block
  // projections from s
  gemm_f32_wmma<<<dim3((32 * 24 + 3) / 4), BLK, 0, stream>>>(s, Wq, bq, q_lin, LQ, 384, 256);
  gemm_f32_wmma<<<dim3((32 * 48 + 3) / 4), BLK, 0, stream>>>(s, Wkv, bkv, kv_lin, LQ, 768, 256);
  gemm_f32_wmma<<<dim3((32 * 6 + 3) / 4), BLK, 0, stream>>>(s, Wqp, bqp, qp_lin, LQ, 96, 256);
  gemm_f32_wmma<<<dim3((32 * 18 + 3) / 4), BLK, 0, stream>>>(s, Wkvp, bkvp, kvp_lin, LQ, 288, 256);
  // fused persistent z-pass (dominant traffic: 134MB read once, bf16 outputs)
  zproj_kernel<<<dim3(ZWAVES / 4), BLK, 0, stream>>>(z, Wb, bb, Wdown, bdown, bias16, pairz);
  // RoPE + rigid transforms + augmented Q'/K' (+ transposed V, V_pts)
  prep_kernel<<<dim3(16), 256, 0, stream>>>(q_lin, kv_lin, qp_lin, kvp_lin, affines,
                                            node_pos, head_w, Qaug, Kaug, VT, VpT);
  // logits + softmax
  logits_kernel<<<dim3(2048), BLK, 0, stream>>>(Qaug, Kaug, bias16, logits);
  softmax_kernel<<<dim3(512), 256, 0, stream>>>(logits, mask, a_bf);
  // o -> feats[:,0:384], o_pt(global) -> optg
  av_kernel<<<dim3(24, 8), BLK, 0, stream>>>(a_bf, VT, feats, 48, 896, 48);
  av_kernel<<<dim3(16, 8), BLK, 0, stream>>>(a_bf, VpT, optg, 24, 192, 24);
  // o_pair -> feats[:,640:896]
  opair_kernel<<<dim3(128), BLK, 0, stream>>>(a_bf, pairz, feats);
  // o_pt inverse frame + norms -> feats[:,384:640]
  optassemble_kernel<<<dim3(128), 256, 0, stream>>>(optg, affines, feats);
  // final projection
  gemm_f32_wmma<<<dim3(128), BLK, 0, stream>>>(feats, Wout, bout, out, LQ, 256, 896);
}
